// CosineSimilarityModel_60017872994998
// MI455X (gfx1250) — matrix operand order repaired and checked by reading, not compiled, not run on hardware
//
#include <hip/hip_runtime.h>

typedef __attribute__((ext_vector_type(16))) _Float16 v16h;
typedef __attribute__((ext_vector_type(8)))  _Float16 v8h;
typedef __attribute__((ext_vector_type(8)))  float    v8f;
typedef __attribute__((ext_vector_type(4)))  float    v4f;
typedef int v4i __attribute__((vector_size(16)));   // b128 payload type

#define DIMK   256
#define NDOC   16384
#define NCLS   32768
#define CTILE  64            // classes per LDS tile (per buffer)
#define WAVES  8
#define BLOCK_ROWS (WAVES * 16)   // 128 doc rows per block
#define CHALF  (NCLS / 2)         // classes per grid.y slice
#define NTILES (CHALF / CTILE)    // 256 tiles per slice
#define CHUNKS_PER_THREAD 8       // 2048 16B chunks / 256 threads

// ---- CDNA5 async global->LDS path (guarded; sync fallback keeps compile) ----
#if defined(__has_builtin)
#  if __has_builtin(__builtin_amdgcn_global_load_async_to_lds_b128)
#    define HAVE_ASYNC_LDS 1
#  endif
#  if __has_builtin(__builtin_amdgcn_s_wait_asynccnt)
#    define HAVE_WAIT_ASYNC 1
#  endif
#endif

__device__ __forceinline__ void async_copy16(const _Float16* g, _Float16* l) {
#if defined(HAVE_ASYNC_LDS)
  __builtin_amdgcn_global_load_async_to_lds_b128(
      (__attribute__((address_space(1))) v4i*)g,
      (__attribute__((address_space(3))) v4i*)l, 0, 0);
#else
  *(v8h*)l = *(const v8h*)g;
#endif
}

template <int N> __device__ __forceinline__ void wait_async_le() {
#if defined(HAVE_ASYNC_LDS)
#  if defined(HAVE_WAIT_ASYNC)
  __builtin_amdgcn_s_wait_asynccnt(N);
#  else
  asm volatile("s_wait_asynccnt %0" :: "n"(N) : "memory");
#  endif
#endif
}

// ---------------------------------------------------------------------------
// Kernel 1: L2-normalize rows of fp32 matrix, emit f16. One wave per row.
// ---------------------------------------------------------------------------
__global__ void normalize_rows_f16(const float* __restrict__ src,
                                   _Float16* __restrict__ dst, int nrows) {
  int wave = threadIdx.x >> 5;
  int lane = threadIdx.x & 31;
  int row  = blockIdx.x * (blockDim.x >> 5) + wave;
  if (row >= nrows) return;

  const float* p = src + (size_t)row * DIMK + lane * 8;
  v4f x0 = *(const v4f*)(p);
  v4f x1 = *(const v4f*)(p + 4);

  float s = x0[0]*x0[0] + x0[1]*x0[1] + x0[2]*x0[2] + x0[3]*x0[3]
          + x1[0]*x1[0] + x1[1]*x1[1] + x1[2]*x1[2] + x1[3]*x1[3];
  #pragma unroll
  for (int m = 16; m >= 1; m >>= 1) s += __shfl_xor(s, m, 32);

  float scale = 1.0f / fmaxf(sqrtf(s), 1e-12f);

  v8h o;
  o[0] = (_Float16)(x0[0] * scale);  o[1] = (_Float16)(x0[1] * scale);
  o[2] = (_Float16)(x0[2] * scale);  o[3] = (_Float16)(x0[3] * scale);
  o[4] = (_Float16)(x1[0] * scale);  o[5] = (_Float16)(x1[1] * scale);
  o[6] = (_Float16)(x1[2] * scale);  o[7] = (_Float16)(x1[3] * scale);
  *(v8h*)(dst + (size_t)row * DIMK + lane * 8) = o;
}

// ---------------------------------------------------------------------------
// Kernel 2: fused  doc_norm @ cls_norm^T -> rowwise (max, argmax) partials.
// Block: 256 threads / 8 waves, 128 doc rows; grid.y splits the classes in 2.
// B tiles (64 classes x 256 K) double-buffered through LDS in WMMA-fragment
// layout via async global->LDS copies (ASYNCcnt-tracked).
// ---------------------------------------------------------------------------
__device__ __forceinline__ void stage_tile(const _Float16* __restrict__ B,
                                           _Float16* lds, int ctbase, int tid) {
  // 2048 x 16B chunks; chunk c = cs*512 + kk*64 + h*32 + l
  #pragma unroll
  for (int it = 0; it < CHUNKS_PER_THREAD; ++it) {
    int c   = it * 256 + tid;
    int cs  = c >> 9;
    int rem = c & 511;
    int kk  = rem >> 6;
    int q   = rem & 63;
    int h   = q >> 5;
    int l   = q & 31;
    int row  = ctbase + cs * 16 + (l & 15);
    int koff = kk * 32 + (l >> 4) * 8 + h * 16;
    async_copy16(B + (size_t)row * DIMK + koff, lds + c * 8);
  }
}

__global__ void __launch_bounds__(256, 1)
cosmax_wmma(const _Float16* __restrict__ A,   // [NDOC][DIMK] f16, normalized
            const _Float16* __restrict__ B,   // [NCLS][DIMK] f16, normalized
            float* __restrict__ partV,        // [2][NDOC]
            int*   __restrict__ partI) {      // [2][NDOC]
  __shared__ _Float16 ldsB[2][CTILE * DIMK];  // 2 x 32 KB

  const int tid  = threadIdx.x;
  const int wave = tid >> 5;
  const int lane = tid & 31;
  const int half = lane >> 4;
  const int l15  = lane & 15;
  const int rowbase = blockIdx.x * BLOCK_ROWS + wave * 16;
  const int cbase   = blockIdx.y * CHALF;

  // ---- A panel resident in registers: 16 rows x 256 K, ISA A-layout ----
  const _Float16* arow = A + (size_t)(rowbase + l15) * DIMK;
  v16h afrag[8];
  #pragma unroll
  for (int kk = 0; kk < 8; ++kk) {
    v8h lo = *(const v8h*)(arow + kk * 32 + half * 8);
    v8h hi = *(const v8h*)(arow + kk * 32 + 16 + half * 8);
    afrag[kk] = __builtin_shufflevector(lo, hi,
        0,1,2,3,4,5,6,7,8,9,10,11,12,13,14,15);
  }

  float maxv[8];
  int   maxi[8];
  #pragma unroll
  for (int r = 0; r < 8; ++r) { maxv[r] = -3.402823466e38f; maxi[r] = 0; }

  stage_tile(B, ldsB[0], cbase, tid);            // prime buffer 0

  for (int t = 0; t < NTILES; ++t) {
    const int cur = t & 1;
    if (t + 1 < NTILES) {                        // stream next tile while
      stage_tile(B, ldsB[cur ^ 1], cbase + (t + 1) * CTILE, tid);
      wait_async_le<CHUNKS_PER_THREAD>();        // tile t's copies done
    } else {
      wait_async_le<0>();
    }
    if (t + 2 < NTILES)                          // warm GL2 ahead of the
      __builtin_prefetch(B + (size_t)(cbase + (t + 2) * CTILE + (tid & 63)) * DIMK, 0, 3);
    __syncthreads();                             // tile t visible to all waves

    const _Float16* buf = ldsB[cur];
    // ---- 4 class-subtiles as 4 independent WMMA chains per K-step ----
    v8f acc[4];
    #pragma unroll
    for (int cs = 0; cs < 4; ++cs) acc[cs] = (v8f){};
    #pragma unroll
    for (int kk = 0; kk < 8; ++kk) {
      v16h bfrag[4];
      #pragma unroll
      for (int cs = 0; cs < 4; ++cs) {
        const _Float16* rr = buf + ((cs * 8 + kk) * 64) * 8;
        v8h blo = *(const v8h*)(rr + lane * 8);
        v8h bhi = *(const v8h*)(rr + (32 + lane) * 8);
        bfrag[cs] = __builtin_shufflevector(blo, bhi,
            0,1,2,3,4,5,6,7,8,9,10,11,12,13,14,15);
      }
      #pragma unroll
      for (int cs = 0; cs < 4; ++cs) {
        acc[cs] = __builtin_amdgcn_wmma_f32_16x16x32_f16(
                      false, afrag[kk], false, bfrag[cs], (short)0, acc[cs],
                      false, false);
      }
    }
    // fused epilogue: running max/argmax (class idx monotonic per lane,
    // strict '>' keeps first occurrence)
    #pragma unroll
    for (int cs = 0; cs < 4; ++cs) {
      int cls = cbase + t * CTILE + cs * 16 + l15;
      #pragma unroll
      for (int r = 0; r < 8; ++r) {
        float a = acc[cs][r];
        if (a > maxv[r]) { maxv[r] = a; maxi[r] = cls; }
      }
    }
    __syncthreads();                             // done reading buf before reuse
  }

  // ---- reduce across 16 class-lanes of each half; write partials ----
  #pragma unroll
  for (int r = 0; r < 8; ++r) {
    float v = maxv[r];
    int   ix = maxi[r];
    #pragma unroll
    for (int m = 8; m >= 1; m >>= 1) {
      float ov = __shfl_xor(v, m, 32);
      int   oi = __shfl_xor(ix, m, 32);
      if (ov > v || (ov == v && oi < ix)) { v = ov; ix = oi; }
    }
    if (l15 == 0) {
      int row = rowbase + r + half * 8;
      partV[blockIdx.y * NDOC + row] = v;
      partI[blockIdx.y * NDOC + row] = ix;
    }
  }
}

// ---------------------------------------------------------------------------
// Kernel 3: merge the two class-half partials (half 0 has smaller indices,
// so strict '>' keeps argmax-first semantics on ties).
// ---------------------------------------------------------------------------
__global__ void merge_parts(const float* __restrict__ partV,
                            const int* __restrict__ partI,
                            float* __restrict__ out) {
  int r = blockIdx.x * 256 + threadIdx.x;
  float v0 = partV[r], v1 = partV[NDOC + r];
  int   i0 = partI[r], i1 = partI[NDOC + r];
  bool take1 = v1 > v0;
  out[r]        = take1 ? v1 : v0;
  out[NDOC + r] = (float)(take1 ? i1 : i0);
}

// ---------------------------------------------------------------------------
extern "C" void kernel_launch(void* const* d_in, const int* in_sizes, int n_in,
                              void* d_out, int out_size, void* d_ws, size_t ws_size,
                              hipStream_t stream) {
  (void)in_sizes; (void)n_in; (void)out_size; (void)ws_size;
  const float* doc = (const float*)d_in[0];   // [16384, 256] fp32
  const float* cls = (const float*)d_in[1];   // [32768, 256] fp32

  char* ws = (char*)d_ws;
  _Float16* docH  = (_Float16*)ws;                                   // 8 MB
  _Float16* clsH  = (_Float16*)(ws + (size_t)NDOC * DIMK * 2);       // 16 MB
  float*    partV = (float*)(ws + (size_t)(NDOC + NCLS) * DIMK * 2); // 128 KB
  int*      partI = (int*)(partV + 2 * NDOC);                        // 128 KB

  normalize_rows_f16<<<NDOC / 8, 256, 0, stream>>>(doc, docH, NDOC);
  normalize_rows_f16<<<NCLS / 8, 256, 0, stream>>>(cls, clsH, NCLS);

  dim3 grid(NDOC / BLOCK_ROWS, 2);
  cosmax_wmma<<<grid, 256, 0, stream>>>(docH, clsH, partV, partI);

  merge_parts<<<NDOC / 256, 256, 0, stream>>>(partV, partI, (float*)d_out);
}
